// OrcDeBERTa_39840116638511
// MI455X (gfx1250) — matrix operand order repaired and checked by reading, not compiled, not run on hardware
//
#include <hip/hip_runtime.h>
#include <hip/hip_bf16.h>
#include <cstdint>

typedef __attribute__((ext_vector_type(16))) _Float16 v16h;
typedef __attribute__((ext_vector_type(8)))  _Float16 v8h;
typedef __attribute__((ext_vector_type(8)))  float    v8f;

#define S_LEN 300
#define SP    304          // 19 * 16
#define INV_SCALE 0.1020620726159657f   // 1/sqrt(3*32)

__device__ __forceinline__ float gelu_f(float x) {
    return 0.5f * x * (1.0f + erff(x * 0.70710678118654752f));
}

__device__ __forceinline__ v16h cat8(v8h x, v8h y) {
    return __builtin_shufflevector(x, y, 0,1,2,3,4,5,6,7,8,9,10,11,12,13,14,15);
}
// B-fragment: 16 contiguous halves (K = koff+0..15 of one column / row-vector)
__device__ __forceinline__ v16h ldh16(const _Float16* p) {
    return cat8(*(const v8h*)p, *(const v8h*)(p + 8));
}
// A-fragment: 16 contiguous halves (K = aoff+0..15)
__device__ __forceinline__ v16h lda16(const _Float16* p) {
    return cat8(*(const v8h*)p, *(const v8h*)(p + 8));
}

#define WMMA(a, b, c) __builtin_amdgcn_wmma_f32_16x16x32_f16(false, (a), false, (b), (short)0, (c), false, false)

// =====================================================================
// WMMA GEMM, f16 operands / f32 accumulate.
// A [M,K] row-major f16;  Wt [N,K] f16 (pre-transposed weights).
// Workgroup = 4 waves covering 64 rows x 32 cols. The 32-column weight
// panel (contiguous 32*K halves in Wt) is copied once per workgroup into
// LDS with GLOBAL_LOAD_ASYNC_TO_LDS_B128 (ASYNCcnt), then all 4 waves
// stream B-fragments from LDS while A streams from global with prefetch.
// Epilogue: +bias (f32), optional erf-GELU, +residual (f32),
// stores fp32 C and/or f16 Ch (each nullable).
// =====================================================================
__global__ __launch_bounds__(128) void gemm_kernel(
    const _Float16* __restrict__ A, const _Float16* __restrict__ Wt,
    const float* __restrict__ bias, const float* __restrict__ R,
    float* __restrict__ C, _Float16* __restrict__ Ch,
    int M, int N, int K, int act)
{
    __shared__ _Float16 s_w[32 * 512];   // 32 KB weight panel (K <= 512)

    int tid  = threadIdx.x;
    int lane = tid & 31;
    int wv   = tid >> 5;
    int j16  = lane & 15;
    int tilesN = (N + 31) >> 5;
    int mb = blockIdx.x / tilesN;        // 64-row macro block
    int n0 = (blockIdx.x % tilesN) << 5;

    // ---- cooperative async copy: weight panel -> LDS ----
    int cols = N - n0; if (cols > 32) cols = 32;
    int chunks = (cols * K) >> 3;        // 8 halves (16B) per chunk
    const _Float16* src = Wt + (size_t)n0 * K;
    for (int i = tid; i < chunks; i += 128) {
        unsigned lds = (unsigned)(uintptr_t)(&s_w[i * 8]);
        const _Float16* g = src + (size_t)i * 8;
        asm volatile("global_load_async_to_lds_b128 %0, %1, off"
                     :: "v"(lds), "v"(g) : "memory");
    }
    asm volatile("s_wait_asynccnt 0" ::: "memory");
    __syncthreads();

    // ---- per-wave 16x32 tile ----
    int m0 = (mb << 6) + (wv << 4);
    int arow = m0 + j16; if (arow >= M) arow = M - 1;
    int c0 = n0 + j16;      if (c0 >= N) c0 = N - 1;  c0 -= n0;
    int c1 = n0 + 16 + j16; if (c1 >= N) c1 = N - 1;  c1 -= n0;

    int koff = (lane < 16) ? 0 : 16;
    const _Float16* ap  = A + (size_t)arow * K + ((lane < 16) ? 0 : 16);
    const _Float16* lw0 = &s_w[c0 * K + koff];
    const _Float16* lw1 = &s_w[c1 * K + koff];

    v8f acc0 = {}, acc1 = {};
    for (int k0 = 0; k0 < K; k0 += 32) {
        if (k0 + 32 < K) __builtin_prefetch(ap + 32, 0, 1);   // global_prefetch_b8
        v16h a = lda16(ap);
        acc0 = WMMA(a, ldh16(lw0), acc0);
        acc1 = WMMA(a, ldh16(lw1), acc1);
        ap += 32; lw0 += 32; lw1 += 32;
    }

    int drow = m0 + ((lane >> 4) << 3);
    #pragma unroll
    for (int vi = 0; vi < 8; ++vi) {
        int m = drow + vi;
        if (m >= M) continue;
        #pragma unroll
        for (int hf = 0; hf < 2; ++hf) {
            int col = n0 + hf * 16 + j16;
            if (col >= N) continue;
            float v = hf ? acc1[vi] : acc0[vi];
            if (bias) v += bias[col];
            if (act == 1) v = gelu_f(v);
            if (R) v += R[(size_t)m * N + col];
            if (C)  C [(size_t)m * N + col] = v;
            if (Ch) Ch[(size_t)m * N + col] = (_Float16)v;
        }
    }
}

// weight convert + transpose: W[K,N] f32 -> Wt[N,K] f16
__global__ void wt_kernel(const float* __restrict__ W, _Float16* __restrict__ Wt,
                          int K, int N)
{
    int idx = blockIdx.x * blockDim.x + threadIdx.x;
    if (idx >= K * N) return;
    int k = idx / N, n = idx % N;
    Wt[(size_t)n * K + k] = (_Float16)W[idx];
}

// plain f32 -> f16 convert
__global__ void cvt_kernel(const float* __restrict__ in, _Float16* __restrict__ out, int n)
{
    int i = blockIdx.x * blockDim.x + threadIdx.x;
    if (i < n) out[i] = (_Float16)in[i];
}

// embed layer 0: [M,2] @ [2,64] + b, GELU -> f16 (K=2 too small for WMMA)
__global__ void embed0_kernel(const float* __restrict__ in, const float* __restrict__ w,
                              const float* __restrict__ b, _Float16* __restrict__ out, int M)
{
    int idx = blockIdx.x * blockDim.x + threadIdx.x;
    if (idx >= M * 64) return;
    int r = idx >> 6, j = idx & 63;
    float v = in[r * 2] * w[j] + in[r * 2 + 1] * w[64 + j] + b[j];
    out[idx] = (_Float16)gelu_f(v);
}

// =====================================================================
// LayerNorm over H=128 (biased var, eps=1e-7). Optional pos-emb add
// before stats, optional mask multiply after. Dual f32 + f16 output.
// =====================================================================
__global__ __launch_bounds__(128) void ln_kernel(
    const float* __restrict__ Y, const float* __restrict__ pos,
    const float* __restrict__ g, const float* __restrict__ bt,
    const int* __restrict__ mask, float* __restrict__ X,
    _Float16* __restrict__ Xh, int M, int S)
{
    int lane = threadIdx.x & 31;
    int row  = blockIdx.x * 4 + (threadIdx.x >> 5);
    if (row >= M) return;
    const float* yp = Y + (size_t)row * 128;
    const float* pp = pos ? pos + (size_t)(row % S) * 128 : nullptr;
    float v[4];
    float s = 0.f, ss = 0.f;
    #pragma unroll
    for (int e = 0; e < 4; ++e) {
        int c = lane + 32 * e;
        float t = yp[c];
        if (pp) t += pp[c];
        v[e] = t; s += t; ss += t * t;
    }
    #pragma unroll
    for (int o = 16; o > 0; o >>= 1) {
        s  += __shfl_xor(s,  o, 32);
        ss += __shfl_xor(ss, o, 32);
    }
    float mean = s * (1.0f / 128.0f);
    float var  = ss * (1.0f / 128.0f) - mean * mean;
    float rstd = rsqrtf(var + 1e-7f);
    float mf = mask ? (mask[row] != 0 ? 1.0f : 0.0f) : 1.0f;
    float*     xp = X  + (size_t)row * 128;
    _Float16*  hp = Xh + (size_t)row * 128;
    #pragma unroll
    for (int e = 0; e < 4; ++e) {
        int c = lane + 32 * e;
        float o = ((v[e] - mean) * rstd * g[c] + bt[c]) * mf;
        xp[c] = o;
        hp[c] = (_Float16)o;
    }
}

// combined qkv bias: [q_bias | 0 | v_bias]
__global__ void qkv_bias_kernel(const float* __restrict__ qb,
                                const float* __restrict__ vb, float* __restrict__ out)
{
    int j = threadIdx.x;
    float v = 0.f;
    if (j < 128) v = qb[j];
    else if (j >= 256) v = vb[j - 256];
    out[j] = v;
}

// =====================================================================
// Fused DeBERTa disentangled attention (all-f16 operands).
// One wave per (b, h, q-chunk16).
// scores = (Q.K^T + c2p + p2c) * inv_scale; XSoftmax; ctx = P @ V.
// c2p(q,k) = Q[q].PosK[q-k+300]; p2c(q,k) = K[k].PosQ[q-k+300].
// Per 16x16 tile r = q-k+300 spans [r0, r0+30], r0 = q0-k0+285:
// two band WMMAs cover it; diagonal gather via LDS.
// =====================================================================
__global__ __launch_bounds__(32) void attn_kernel(
    const _Float16* __restrict__ QKV,   // [B*S, 384] f16
    const _Float16* __restrict__ PK,    // [600, 128] f16 pos_k proj
    const _Float16* __restrict__ PQ,    // [600, 128] f16 pos_q proj (+bias)
    const int*      __restrict__ mask,  // [B, S]
    _Float16* __restrict__ CTX)         // [B*S, 128] f16
{
    __shared__ float    s_sc[16][SP + 8];
    __shared__ _Float16 s_pb[16][320];
    __shared__ float    s_band[16][32];

    int lane = threadIdx.x;
    int qc = blockIdx.x, h = blockIdx.y, b = blockIdx.z;
    int q0 = qc * 16;
    size_t rowb = (size_t)b * S_LEN;
    int cq = h * 32, ck = 128 + h * 32, cv = 256 + h * 32, ch = h * 32;

    int aoff = (lane < 16) ? 0 : 16;
    int koff = (lane < 16) ? 0 : 16;
    int j16  = lane & 15;
    int dr   = (lane >> 4) << 3;

    // Q A-fragment (DH=32 == one WMMA K-step)
    v16h qa;
    {
        int qr = q0 + j16; if (qr > S_LEN - 1) qr = S_LEN - 1;
        qa = lda16(QKV + (rowb + qr) * 384 + cq + aoff);
    }

    for (int t = 0; t < 19; ++t) {
        int k0 = t * 16;
        int r0 = q0 - k0 + 285;

        int kr = k0 + j16; if (kr > S_LEN - 1) kr = S_LEN - 1;
        int ra = r0 + j16;      if (ra < 0) ra = 0; if (ra > 599) ra = 599;
        int rb = r0 + 16 + j16; if (rb < 0) rb = 0; if (rb > 599) rb = 599;

        // ---- Q.K^T ----
        v16h kb = ldh16(QKV + (rowb + kr) * 384 + ck + koff);
        v8f acc = {};
        acc = WMMA(qa, kb, acc);

        // ---- c2p band: Q @ PosK[r0 .. r0+31]^T ----
        v8f b1 = {}, b2 = {};
        b1 = WMMA(qa, ldh16(PK + (size_t)ra * 128 + ch + koff), b1);
        b2 = WMMA(qa, ldh16(PK + (size_t)rb * 128 + ch + koff), b2);
        #pragma unroll
        for (int vi = 0; vi < 8; ++vi) {
            s_band[dr + vi][j16]      = b1[vi];
            s_band[dr + vi][16 + j16] = b2[vi];
        }
        __syncthreads();
        #pragma unroll
        for (int vi = 0; vi < 8; ++vi) {
            int qq = dr + vi;
            acc[vi] += s_band[qq][qq - j16 + 15];
        }
        __syncthreads();

        // ---- p2c band: K_tile @ PosQ[r0 .. r0+31]^T ----
        v16h ka = lda16(QKV + (rowb + kr) * 384 + ck + aoff);
        v8f b3 = {}, b4 = {};
        b3 = WMMA(ka, ldh16(PQ + (size_t)ra * 128 + ch + koff), b3);
        b4 = WMMA(ka, ldh16(PQ + (size_t)rb * 128 + ch + koff), b4);
        #pragma unroll
        for (int vi = 0; vi < 8; ++vi) {
            s_band[dr + vi][j16]      = b3[vi];
            s_band[dr + vi][16 + j16] = b4[vi];
        }
        __syncthreads();
        #pragma unroll
        for (int vi = 0; vi < 8; ++vi) {
            int qq = dr + vi;
            acc[vi] += s_band[j16][qq - j16 + 15];   // band2[kk][qq-kk+15]
        }
        __syncthreads();

        // ---- scale + key mask + stage scores ----
        #pragma unroll
        for (int vi = 0; vi < 8; ++vi) {
            int qq = dr + vi;
            int kidx = k0 + j16;
            bool ok = (kidx < S_LEN) && (mask[rowb + kidx] != 0);
            s_sc[qq][k0 + j16] = ok ? acc[vi] * INV_SCALE : -3.0e38f;
        }
        __syncthreads();
    }

    // ---- XSoftmax over 304 staged columns (lanes 0..15 = rows) ----
    if (lane < 16) {
        int q = q0 + lane;
        float mq = (q < S_LEN && mask[rowb + q] != 0) ? 1.0f : 0.0f;
        float mx = -3.0e38f;
        for (int c = 0; c < SP; ++c) mx = fmaxf(mx, s_sc[lane][c]);
        float sum = 0.f;
        for (int c = 0; c < SP; ++c) {
            float e = expf(s_sc[lane][c] - mx);
            s_sc[lane][c] = e;
            sum += e;
        }
        float rn = mq / sum;
        for (int c = 0; c < SP; ++c) {
            float mk = (c < S_LEN && mask[rowb + c] != 0) ? 1.0f : 0.0f;
            s_pb[lane][c] = (_Float16)(s_sc[lane][c] * rn * mk);
        }
        for (int c = SP; c < 320; ++c) s_pb[lane][c] = (_Float16)0.0f;
    }
    __syncthreads();

    // ---- ctx = P @ V  (10 K-steps of 32) ----
    v8f c0 = {}, c1 = {};
    for (int t = 0; t < 10; ++t) {
        int kk0 = t * 32;
        v16h pa;
        {
            int poff = kk0 + aoff;
            #pragma unroll
            for (int j = 0; j < 8; ++j) {
                pa[j]     = s_pb[j16][poff + j];
                pa[8 + j] = s_pb[j16][poff + 8 + j];
            }
        }
        v16h vb0, vb1;
        #pragma unroll
        for (int j = 0; j < 16; ++j) {
            int r = kk0 + koff + j; if (r > S_LEN - 1) r = S_LEN - 1;
            const _Float16* vp = QKV + (rowb + r) * 384 + cv;
            vb0[j] = vp[j16];
            vb1[j] = vp[16 + j16];
        }
        c0 = WMMA(pa, vb0, c0);
        c1 = WMMA(pa, vb1, c1);
    }
    #pragma unroll
    for (int vi = 0; vi < 8; ++vi) {
        int q = q0 + dr + vi;
        if (q < S_LEN) {
            _Float16* op = CTX + (rowb + q) * 128 + h * 32;
            op[j16]      = (_Float16)c0[vi];
            op[16 + j16] = (_Float16)c1[vi];
        }
    }
}

// =====================================================================
// host orchestration
// =====================================================================
static inline void launch_gemm(const _Float16* A, const _Float16* Wt,
                               const float* bias, const float* R,
                               float* C, _Float16* Ch,
                               int M, int N, int K, int act, hipStream_t s)
{
    int blocks = ((M + 63) / 64) * ((N + 31) / 32);
    gemm_kernel<<<blocks, 128, 0, s>>>(A, Wt, bias, R, C, Ch, M, N, K, act);
}

extern "C" void kernel_launch(void* const* d_in, const int* in_sizes, int n_in,
                              void* d_out, int out_size, void* d_ws, size_t ws_size,
                              hipStream_t stream)
{
    (void)in_sizes; (void)n_in; (void)out_size; (void)ws_size;
    const float* inputs   = (const float*)d_in[0];
    const int*   mask     = (const int*)  d_in[1];
    const float* e0_w     = (const float*)d_in[2];
    const float* e0_b     = (const float*)d_in[3];
    const float* e1_w     = (const float*)d_in[4];
    const float* e1_b     = (const float*)d_in[5];
    const float* e2_w     = (const float*)d_in[6];
    const float* e2_b     = (const float*)d_in[7];
    const float* pos_emb  = (const float*)d_in[8];
    const float* eln_g    = (const float*)d_in[9];
    const float* eln_b    = (const float*)d_in[10];
    const float* rel_emb  = (const float*)d_in[11];
    const float* in_w     = (const float*)d_in[12];
    const float* q_bias   = (const float*)d_in[13];
    const float* v_bias   = (const float*)d_in[14];
    const float* pos_k_w  = (const float*)d_in[15];
    const float* pos_q_w  = (const float*)d_in[16];
    const float* pos_q_b  = (const float*)d_in[17];
    const float* o_w      = (const float*)d_in[18];
    const float* o_b      = (const float*)d_in[19];
    const float* ln1_g    = (const float*)d_in[20];
    const float* ln1_b    = (const float*)d_in[21];
    const float* ffn_i_w  = (const float*)d_in[22];
    const float* ffn_i_b  = (const float*)d_in[23];
    const float* ffn_o_w  = (const float*)d_in[24];
    const float* ffn_o_b  = (const float*)d_in[25];
    const float* ln2_g    = (const float*)d_in[26];
    const float* ln2_b    = (const float*)d_in[27];
    const float* r0_w     = (const float*)d_in[28];
    const float* r0_b     = (const float*)d_in[29];
    const float* r1_w     = (const float*)d_in[30];
    const float* r1_b     = (const float*)d_in[31];
    const float* r2_w     = (const float*)d_in[32];
    const float* r2_b     = (const float*)d_in[33];

    const int M = 128 * S_LEN;   // 38400 rows

    char* base = (char*)d_ws;
    size_t off = 0;
    auto allocB = [&](size_t bytes) -> void* {
        off = (off + 255) & ~(size_t)255;
        void* p = base + off;
        off += bytes;
        return p;
    };
    float*    X    = (float*)   allocB((size_t)M * 128 * 4);  // residual stream f32
    float*    Yb   = (float*)   allocB((size_t)M * 128 * 4);  // pre-LN scratch f32
    _Float16* Xh   = (_Float16*)allocB((size_t)M * 128 * 2);  // f16 residual copy / E1h
    _Float16* CTXh = (_Float16*)allocB((size_t)M * 128 * 2);  // attn ctx / recon h1
    _Float16* BIGh = (_Float16*)allocB((size_t)M * 512 * 2);  // E0h/QKVh/HFFh/recon h2
    _Float16* PKh  = (_Float16*)allocB((size_t)600 * 128 * 2);
    _Float16* PQh  = (_Float16*)allocB((size_t)600 * 128 * 2);
    _Float16* REh  = (_Float16*)allocB((size_t)600 * 128 * 2); // rel_emb f16
    float*    BQ   = (float*)   allocB(384 * 4);
    // f16 transposed weights
    _Float16* e1t  = (_Float16*)allocB((size_t)64 * 128 * 2);
    _Float16* e2t  = (_Float16*)allocB((size_t)128 * 128 * 2);
    _Float16* inwt = (_Float16*)allocB((size_t)8 * 128 * 384 * 2);
    _Float16* pkwt = (_Float16*)allocB((size_t)8 * 128 * 128 * 2);
    _Float16* pqwt = (_Float16*)allocB((size_t)8 * 128 * 128 * 2);
    _Float16* owt  = (_Float16*)allocB((size_t)8 * 128 * 128 * 2);
    _Float16* fiwt = (_Float16*)allocB((size_t)8 * 128 * 512 * 2);
    _Float16* fowt = (_Float16*)allocB((size_t)8 * 512 * 128 * 2);
    _Float16* r0t  = (_Float16*)allocB((size_t)128 * 128 * 2);
    _Float16* r1t  = (_Float16*)allocB((size_t)128 * 64 * 2);
    _Float16* r2t  = (_Float16*)allocB((size_t)64 * 5 * 2);

    auto wt = [&](const float* W, _Float16* Wt, int K, int N) {
        int n = K * N;
        wt_kernel<<<(n + 255) / 256, 256, 0, stream>>>(W, Wt, K, N);
    };

    // ---- one-time weight / table conversions ----
    wt(e1_w, e1t, 64, 128);
    wt(e2_w, e2t, 128, 128);
    for (int l = 0; l < 8; ++l) {
        wt(in_w    + (size_t)l * 128 * 384, inwt + (size_t)l * 128 * 384, 128, 384);
        wt(pos_k_w + (size_t)l * 128 * 128, pkwt + (size_t)l * 128 * 128, 128, 128);
        wt(pos_q_w + (size_t)l * 128 * 128, pqwt + (size_t)l * 128 * 128, 128, 128);
        wt(o_w     + (size_t)l * 128 * 128, owt  + (size_t)l * 128 * 128, 128, 128);
        wt(ffn_i_w + (size_t)l * 128 * 512, fiwt + (size_t)l * 128 * 512, 128, 512);
        wt(ffn_o_w + (size_t)l * 512 * 128, fowt + (size_t)l * 512 * 128, 512, 128);
    }
    wt(r0_w, r0t, 128, 128);
    wt(r1_w, r1t, 128, 64);
    wt(r2_w, r2t, 64, 5);
    cvt_kernel<<<(600 * 128 + 255) / 256, 256, 0, stream>>>(rel_emb, REh, 600 * 128);

    // ---- embedding MLP 2 -> 64 -> 128 -> 128, +pos, LN, mask ----
    embed0_kernel<<<(M * 64 + 255) / 256, 256, 0, stream>>>(inputs, e0_w, e0_b, BIGh, M);
    launch_gemm(BIGh, e1t, e1_b, nullptr, nullptr, Xh, M, 128, 64, 1, stream);   // E1h in Xh
    launch_gemm(Xh, e2t, e2_b, nullptr, Yb, nullptr, M, 128, 128, 0, stream);
    ln_kernel<<<M / 4, 128, 0, stream>>>(Yb, pos_emb, eln_g, eln_b, mask, X, Xh, M, S_LEN);

    // ---- 8 DeBERTa layers ----
    for (int l = 0; l < 8; ++l) {
        qkv_bias_kernel<<<1, 384, 0, stream>>>(q_bias + l * 128, v_bias + l * 128, BQ);
        launch_gemm(Xh, inwt + (size_t)l * 128 * 384, BQ, nullptr, nullptr, BIGh, M, 384, 128, 0, stream);
        launch_gemm(REh, pkwt + (size_t)l * 128 * 128, nullptr, nullptr, nullptr, PKh, 600, 128, 128, 0, stream);
        launch_gemm(REh, pqwt + (size_t)l * 128 * 128, pos_q_b + l * 128, nullptr, nullptr, PQh, 600, 128, 128, 0, stream);
        attn_kernel<<<dim3(19, 4, 128), 32, 0, stream>>>(BIGh, PKh, PQh, mask, CTXh);
        launch_gemm(CTXh, owt + (size_t)l * 128 * 128, o_b + l * 128, X, Yb, nullptr, M, 128, 128, 0, stream);
        ln_kernel<<<M / 4, 128, 0, stream>>>(Yb, nullptr, ln1_g + l * 128, ln1_b + l * 128, nullptr, X, Xh, M, S_LEN);
        launch_gemm(Xh, fiwt + (size_t)l * 128 * 512, ffn_i_b + l * 512, nullptr, nullptr, BIGh, M, 512, 128, 1, stream);
        launch_gemm(BIGh, fowt + (size_t)l * 512 * 128, ffn_o_b + l * 128, X, Yb, nullptr, M, 128, 512, 0, stream);
        ln_kernel<<<M / 4, 128, 0, stream>>>(Yb, nullptr, ln2_g + l * 128, ln2_b + l * 128, nullptr, X, Xh, M, S_LEN);
    }

    // ---- reconstruction head 128 -> 128 -> 64 -> 5 ----
    launch_gemm(Xh, r0t, r0_b, nullptr, nullptr, CTXh, M, 128, 128, 1, stream);
    launch_gemm(CTXh, r1t, r1_b, nullptr, nullptr, BIGh, M, 64, 128, 1, stream);
    launch_gemm(BIGh, r2t, r2_b, nullptr, (float*)d_out, nullptr, M, 5, 64, 0, stream);
}
